// DynamicConv1d_3272765079874
// MI455X (gfx1250) — matrix operand order repaired and checked by reading, not compiled, not run on hardware
//
#include <hip/hip_runtime.h>
#include <hip/hip_bf16.h>

// ---- problem constants (from reference) ----
#define NB      8
#define CH      1024
#define TLEN    2048
#define HEADS   16
#define KS      7
#define HK      112        // HEADS*KS
#define RCH     64         // CH/HEADS
#define PADL    6

// ---- GEMM tiling ----
#define TM      16         // time-tile (M)
#define KT      32         // K step (f16 wmma depth)
#define KSTEPS  (CH / KT)  // 32
#define NTILES  7          // 112 / 16
#define CP      1032       // padded LDS row stride in halves (16*stride covers all 64 banks)

typedef __attribute__((ext_vector_type(16))) _Float16 v16h;
typedef __attribute__((ext_vector_type(8)))  float    v8f;

// ============================================================================
// Kernel 1: pack Wg [112, 1024] f32 into exact per-lane B-fragment images.
// For V_WMMA_F32_16X16X32_F16, B (32x16) per-lane layout (ISA 7.12.2):
//   lane n%16 = column N; lanes 0-15 hold K=0..7,16..23 (2 per VGPR), lanes 16-31 K=8..15,24..31.
// bf index: (((kt*NTILES + nt)*32 + lane)*16 + i), i = half index (v=i/2, h=i%2).
// ============================================================================
__global__ __launch_bounds__(256) void pack_b_kernel(const float* __restrict__ Wg,
                                                     _Float16* __restrict__ bf) {
    int idx = blockIdx.x * 256 + threadIdx.x;
    const int total = KSTEPS * NTILES * 32 * 16;     // 114688
    if (idx >= total) return;
    int i    = idx & 15;
    int lane = (idx >> 4) & 31;
    int nt   = (idx >> 9) % NTILES;
    int kt   = idx / (16 * 32 * NTILES);
    int v = i >> 1, h = i & 1;
    int K = ((2 * v + h) & 7) + ((v >= 4) ? 16 : 0) + ((lane >= 16) ? 8 : 0);
    int c = kt * KT + K;
    int o = nt * 16 + (lane & 15);                   // output column = row of Wg
    bf[idx] = (_Float16)Wg[(size_t)o * CH + c];
}

// ============================================================================
// Kernel 2: logits GEMM (WMMA f16->f32) + bias + softmax over K=7 per head.
// Grid: NB * (TLEN/TM) blocks, 128 threads (4 waves).
// ============================================================================
__global__ __launch_bounds__(128) void logits_softmax_kernel(
    const float* __restrict__ x,        // [B, C, T]
    const v16h*  __restrict__ bfr,      // packed B fragments
    const float* __restrict__ bg,       // [112]
    float*       __restrict__ wout)     // [B, T, H, K] softmax weights
{
    __shared__ _Float16 xs[TM * CP];          // x tile, f16, xs[m*CP + c]
    __shared__ float    lbuf[TM * HK];        // logits tile

    const int tid = threadIdx.x;
    const int blk = blockIdx.x;
    const int b   = blk / (TLEN / TM);
    const int t0  = (blk % (TLEN / TM)) * TM;

    // ---- stage x[b][:, t0:t0+16] into LDS as f16 (coalesced: 16 threads/channel)
    {
        int m = tid & 15;
        int cstart = tid >> 4;                           // 0..7
        const float* xb = x + (size_t)b * CH * TLEN + t0 + m;
        for (int c = cstart; c < CH; c += 8)
            xs[m * CP + c] = (_Float16)xb[(size_t)c * TLEN];
    }
    __syncthreads();

    const int lane = tid & 31;
    const int wv   = tid >> 5;
    const int m    = lane & 15;
    const int hi   = lane >> 4;                          // 0/1: lane half

    // Each wave owns n-tiles {wv, wv+4} (wave 3 owns only tile 3): 7 tiles total.
    for (int sel = 0; sel < 2; ++sel) {
        int nt = wv + sel * 4;
        if (nt >= NTILES) break;                         // wave-uniform branch
        v8f acc = {0.f, 0.f, 0.f, 0.f, 0.f, 0.f, 0.f, 0.f};

        for (int kt = 0; kt < KSTEPS; ++kt) {
            const int c0 = kt * KT;
            // A fragment, 16x32 f16 (ISA 7.12.2): lane m%16 = row M;
            // VGPR v: K = (v%4)*2 + (v>=4?16:0) + (lane>=16?8:0), halves = K,K+1.
            union { v16h v; unsigned u[8]; } au;
            #pragma unroll
            for (int v = 0; v < 8; ++v) {
                int kA = ((v & 3) * 2) + ((v >= 4) ? 16 : 0) + hi * 8;
                au.u[v] = *(const unsigned*)&xs[m * CP + c0 + kA];  // 2 halves at once
            }
            // B fragment: one coalesced 32B load per lane (pre-packed image)
            v16h bb = bfr[(size_t)(kt * NTILES + nt) * 32 + lane];
            acc = __builtin_amdgcn_wmma_f32_16x16x32_f16(
                      false, au.v, false, bb, (short)0, acc, false, false);
        }
        // C/D layout: VGPR j -> (M = j + 8*hi, N = nt*16 + m)
        #pragma unroll
        for (int j = 0; j < 8; ++j)
            lbuf[(j + hi * 8) * HK + nt * 16 + m] = acc[j];
    }
    __syncthreads();

    // ---- softmax over K=7 taps: 16 t x 16 h = 256 tasks on 128 threads
    for (int task = tid; task < TM * HEADS; task += 128) {
        int h = task & 15;
        int t = task >> 4;
        float lg[KS];
        float mx = -1e30f;
        #pragma unroll
        for (int k = 0; k < KS; ++k) {
            lg[k] = lbuf[t * HK + h * KS + k] + bg[h * KS + k];
            mx = fmaxf(mx, lg[k]);
        }
        float s = 0.f;
        #pragma unroll
        for (int k = 0; k < KS; ++k) { lg[k] = __expf(lg[k] - mx); s += lg[k]; }
        float inv = 1.0f / s;
        float* wo = wout + (((size_t)b * TLEN + t0 + t) * HEADS + h) * KS;
        #pragma unroll
        for (int k = 0; k < KS; ++k) wo[k] = lg[k] * inv;
    }
}

// ============================================================================
// Kernel 3: 7-tap causal weighted sum + per-head bias.
// Grid: (B*H, T/256), 256 threads. Each thread: fixed t, loops 64 head channels.
// x is L2-resident (64 MB < 192 MB), weights cached in registers across r-loop.
// ============================================================================
__global__ __launch_bounds__(256) void conv_kernel(
    const float* __restrict__ x,        // [B, C, T]
    const float* __restrict__ wsm,      // [B, T, H, K]
    const float* __restrict__ bias,     // [H]
    float*       __restrict__ out)      // [B, C, T]
{
    int bh = blockIdx.x;
    int b = bh >> 4, h = bh & 15;
    int t = blockIdx.y * 256 + threadIdx.x;

    float wk[KS];
    const float* wp = wsm + (((size_t)b * TLEN + t) * HEADS + h) * KS;
    #pragma unroll
    for (int k = 0; k < KS; ++k) wk[k] = wp[k];
    float bs = bias[h];

    const float* xb = x   + ((size_t)b * CH + h * RCH) * TLEN;
    float*       ob = out + ((size_t)b * CH + h * RCH) * TLEN;

    for (int r = 0; r < RCH; ++r) {
        const float* xr = xb + (size_t)r * TLEN;
        float acc = bs;
        #pragma unroll
        for (int k = 0; k < KS; ++k) {
            int ts = t - PADL + k;                   // causal: left pad 6, no right pad
            float xv = (ts >= 0) ? xr[ts] : 0.0f;
            acc = fmaf(wk[k], xv, acc);
        }
        ob[(size_t)r * TLEN + t] = acc;
    }
}

// ============================================================================
// Launch
// ============================================================================
extern "C" void kernel_launch(void* const* d_in, const int* in_sizes, int n_in,
                              void* d_out, int out_size, void* d_ws, size_t ws_size,
                              hipStream_t stream) {
    const float* x    = (const float*)d_in[0];   // [8,1024,2048]
    const float* Wg   = (const float*)d_in[1];   // [112,1024]
    const float* bg   = (const float*)d_in[2];   // [112]
    const float* bias = (const float*)d_in[3];   // [16]
    float* out = (float*)d_out;

    // workspace layout: [0, 256KB) packed B frags (f16, 224 KB); then softmax w (f32, 7 MB)
    _Float16* bf  = (_Float16*)d_ws;
    float*    wsm = (float*)((char*)d_ws + (256u << 10));

    const int bf_elems = KSTEPS * NTILES * 32 * 16;               // 114688
    pack_b_kernel<<<(bf_elems + 255) / 256, 256, 0, stream>>>(Wg, bf);

    logits_softmax_kernel<<<NB * (TLEN / TM), 128, 0, stream>>>(
        x, (const v16h*)bf, bg, wsm);

    dim3 cgrid(NB * HEADS, TLEN / 256);
    conv_kernel<<<cgrid, 256, 0, stream>>>(x, wsm, bias, out);
}
